// RandLANet_54580444398212
// MI455X (gfx1250) — matrix operand order repaired and checked by reading, not compiled, not run on hardware
//
#include <hip/hip_runtime.h>
#include <hip/hip_bf16.h>

// ---------------------------------------------------------------------------
// RandLA-Net forward for MI455X (gfx1250, wave32, WMMA).
// ---------------------------------------------------------------------------

typedef __attribute__((ext_vector_type(16))) _Float16 v16h;
typedef __attribute__((ext_vector_type(8)))  float    v8f;

// ---------------------------------------------------------------------------
// Weight pre-pack: W (O x C, f32, row-major) -> zero-padded f16 fragment
// layout [tileO][chunk][lane 0..31][j 0..15] with k = 16*(lane/16) + j, the
// contiguous k order the B loads also use: the GEMM loads one contiguous v16h per
// lane per 32-channel chunk.
// ---------------------------------------------------------------------------
__global__ void pack_w_kernel(const float* __restrict__ W,
                              _Float16* __restrict__ Wp,
                              int O, int C, int nch, long long total)
{
    const long long i = (long long)blockIdx.x * 256 + threadIdx.x;
    if (i >= total) return;
    const int j    = (int)(i & 15);
    const int lane = (int)((i >> 4) & 31);
    const long long rest = i >> 9;
    const int ch = (int)(rest % nch);
    const int tO = (int)(rest / nch);
    const int m = lane & 15, hi = lane >> 4;
    const int v = j >> 1, s = j & 1;
    const int k = 2 * v + s + 16 * hi;
    const int oi = tO * 16 + m, ci = ch * 32 + k;
    const float w = (oi < O && ci < C) ? W[(long long)oi * C + ci] : 0.0f;
    Wp[i] = (_Float16)w;
}

// ---------------------------------------------------------------------------
// WMMA GEMM: Y[o,p] = epilogue( sum_c W[o,c] * X[c,p] ), X channel-major
// [Cpad][P], zero-padded rows, P % 16 == 0. Each wave computes MT 16x16
// output tiles along O that SHARE one B fragment per 32-chunk: B is 16
// coalesced strided loads + cvt, A is one v16h vector load per tile, so MT
// amortizes the activation traffic across MT v_wmma_f32_16x16x32_f16 ops.
// Epilogue fuses bias + BN + residual + ReLU with batched loads.
// ---------------------------------------------------------------------------
template <int MT>
__global__ void __launch_bounds__(256)
wmma_gemm_kernel(const _Float16* __restrict__ Wp, const float* __restrict__ bias,
                 const float* __restrict__ bng, const float* __restrict__ bnbe,
                 const float* __restrict__ X, const float* __restrict__ R,
                 float* __restrict__ Y, int O, int nch, long long P,
                 int do_relu, int out_mode, int Nper)
{
    const int lane = threadIdx.x & 31;
    const int wave = threadIdx.x >> 5;
    const long long tilesP = P >> 4;            // P % 16 == 0 by construction
    const int tilesO = (O + 15) >> 4;
    const int groupsO = tilesO / MT;            // host guarantees divisibility
    const long long tile = (long long)blockIdx.x * 8 + wave;
    if (tile >= (long long)groupsO * tilesP) return;
    const int tG = (int)(tile / tilesP);
    const long long p0 = (tile % tilesP) << 4;

    const int mlane = lane & 15;
    const int hi = lane >> 4;

    const v16h* wp = (const v16h*)Wp + ((long long)(tG * MT) * nch * 32 + lane);
    const float* xcol = X + p0 + mlane;

    v8f acc[MT];
#pragma unroll
    for (int t = 0; t < MT; ++t) acc[t] = (v8f){};

    for (int ch = 0; ch < nch; ++ch) {
        const float* xc = xcol + ((long long)(ch * 32 + hi * 16)) * P;
        v16h b = {};
#pragma unroll
        for (int j = 0; j < 16; ++j)
            b[j] = (_Float16)xc[(long long)j * P];
#pragma unroll
        for (int t = 0; t < MT; ++t) {
            const v16h a = wp[((long long)t * nch + ch) * 32];
            acc[t] = __builtin_amdgcn_wmma_f32_16x16x32_f16(
                false, a, false, b, (short)0, acc[t], false, false);
        }
    }

    const float bnscale = 0.99999950000037496f; // 1/sqrt(1 + 1e-6)
    const long long pi = p0 + mlane;
#pragma unroll
    for (int t = 0; t < MT; ++t) {
        const int o0 = (tG * MT + t) * 16;
        if (o0 + 16 <= O) {                     // uniform: full output tile
            float yv[8], bia[8];
#pragma unroll
            for (int v = 0; v < 8; ++v) bia[v] = bias[o0 + v + 8 * hi];
#pragma unroll
            for (int v = 0; v < 8; ++v) yv[v] = acc[t][v] + bia[v];
            if (bng) {
                float gg[8], bb[8];
#pragma unroll
                for (int v = 0; v < 8; ++v) {
                    gg[v] = bng[o0 + v + 8 * hi];
                    bb[v] = bnbe[o0 + v + 8 * hi];
                }
#pragma unroll
                for (int v = 0; v < 8; ++v) yv[v] = gg[v] * yv[v] * bnscale + bb[v];
            }
            if (R) {
                float rr[8];
#pragma unroll
                for (int v = 0; v < 8; ++v)
                    rr[v] = R[(long long)(o0 + v + 8 * hi) * P + pi];
#pragma unroll
                for (int v = 0; v < 8; ++v) yv[v] += rr[v];
            }
            if (do_relu) {
#pragma unroll
                for (int v = 0; v < 8; ++v) yv[v] = fmaxf(yv[v], 0.0f);
            }
#pragma unroll
            for (int v = 0; v < 8; ++v)
                Y[(long long)(o0 + v + 8 * hi) * P + pi] = yv[v];
        } else {                                // edge tile (O = 8 or 13)
#pragma unroll
            for (int v = 0; v < 8; ++v) {
                const int o = o0 + v + 8 * hi;
                if (o < O) {
                    float y = acc[t][v] + bias[o];
                    if (bng)     y = bng[o] * y * bnscale + bnbe[o];
                    if (R)       y += R[(long long)o * P + pi];
                    if (do_relu) y = fmaxf(y, 0.0f);
                    if (out_mode == 0) {
                        Y[(long long)o * P + pi] = y;
                    } else {
                        const long long bb2 = pi / Nper, n = pi % Nper;
                        Y[(bb2 * O + o) * (long long)Nper + n] = y;
                    }
                }
            }
        }
    }
}

// ---------------------------------------------------------------------------
// Fused pairwise-distance + top-K KNN. One query per lane; refs staged
// through LDS in 256-point tiles; register-resident insertion top-K.
// Never materializes the n x n distance matrix (512MB/batch at n=8192 ->
// ~44us of HBM traffic on its own at 23.3 TB/s).
// ---------------------------------------------------------------------------
template <int KK>
__global__ void __launch_bounds__(256)
knn_kernel(const float* __restrict__ coords, int N_all, int nref, int nq,
           int* __restrict__ idx, float* __restrict__ dist)
{
    const int b = blockIdx.y;
    const int q = blockIdx.x * 256 + threadIdx.x;
    const int qc = (q < nq) ? q : (nq - 1);
    const float* cb = coords + (long long)b * N_all * 3;
    const float qx = cb[qc * 3 + 0];
    const float qy = cb[qc * 3 + 1];
    const float qz = cb[qc * 3 + 2];

    float bd[KK];
    int   bi[KK];
#pragma unroll
    for (int k = 0; k < KK; ++k) { bd[k] = 3.4e38f; bi[k] = 0; }

    __shared__ float sref[256 * 3];
    for (int t0 = 0; t0 < nref; t0 += 256) {
        const int tn = min(256, nref - t0);
        __syncthreads();
        if ((int)threadIdx.x < tn) {
            sref[threadIdx.x * 3 + 0] = cb[(t0 + threadIdx.x) * 3 + 0];
            sref[threadIdx.x * 3 + 1] = cb[(t0 + threadIdx.x) * 3 + 1];
            sref[threadIdx.x * 3 + 2] = cb[(t0 + threadIdx.x) * 3 + 2];
        }
        __syncthreads();
        for (int j = 0; j < tn; ++j) {
            const float dx = qx - sref[j * 3 + 0];
            const float dy = qy - sref[j * 3 + 1];
            const float dz = qz - sref[j * 3 + 2];
            float d2 = dx * dx + dy * dy + dz * dz;
            if (d2 < bd[KK - 1]) {
                int ic = t0 + j;
#pragma unroll
                for (int k = 0; k < KK; ++k) {
                    if (d2 < bd[k]) {
                        float tf = bd[k]; bd[k] = d2; d2 = tf;
                        int   ti = bi[k]; bi[k] = ic; ic = ti;
                    }
                }
            }
        }
    }
    if (q < nq) {
#pragma unroll
        for (int k = 0; k < KK; ++k) {
            idx [((long long)b * nq + q) * KK + k] = bi[k];
            dist[((long long)b * nq + q) * KK + k] = sqrtf(fmaxf(bd[k], 0.0f));
        }
    }
}

// ---------------------------------------------------------------------------
// Elementwise / reshaping kernels.
// ---------------------------------------------------------------------------
__global__ void zerofill_kernel(float* __restrict__ p, long long total)
{
    const long long i = (long long)blockIdx.x * 256 + threadIdx.x;
    if (i < total) p[i] = 0.0f;
}

__global__ void reduce_absmax_kernel(const float* __restrict__ in,
                                     float* __restrict__ rmax, int Npts)
{
    __shared__ float sm[3][256];
    const int b = blockIdx.x, t = threadIdx.x;
    float m0 = 0.f, m1 = 0.f, m2 = 0.f;
    for (int n = t; n < Npts; n += 256) {
        const float* p = in + ((long long)b * Npts + n) * 4;
        m0 = fmaxf(m0, fabsf(p[0]));
        m1 = fmaxf(m1, fabsf(p[1]));
        m2 = fmaxf(m2, fabsf(p[2]));
    }
    sm[0][t] = m0; sm[1][t] = m1; sm[2][t] = m2;
    __syncthreads();
    for (int s = 128; s > 0; s >>= 1) {
        if (t < s) {
            sm[0][t] = fmaxf(sm[0][t], sm[0][t + s]);
            sm[1][t] = fmaxf(sm[1][t], sm[1][t + s]);
            sm[2][t] = fmaxf(sm[2][t], sm[2][t + s]);
        }
        __syncthreads();
    }
    if (t < 3) rmax[b * 3 + t] = sm[t][0];
}

__global__ void normalize_kernel(const float* __restrict__ in,
                                 const float* __restrict__ rmax,
                                 float* __restrict__ coords,
                                 float* __restrict__ xin,
                                 int Npts, long long total)
{
    const long long i = (long long)blockIdx.x * 256 + threadIdx.x;
    if (i >= total) return;
    const int b = (int)(i / Npts);
    const float* p = in + i * 4;
    const float x = p[0] / (rmax[b * 3 + 0] + 1e-6f);
    const float y = p[1] / (rmax[b * 3 + 1] + 1e-6f);
    const float z = p[2] / (rmax[b * 3 + 2] + 1e-6f);
    coords[i * 3 + 0] = x; coords[i * 3 + 1] = y; coords[i * 3 + 2] = z;
    xin[0 * total + i] = x;
    xin[1 * total + i] = y;
    xin[2 * total + i] = z;
    xin[3 * total + i] = p[3] - 0.5f;
}

// Build the 10-channel LSE tensor: [ctr(3), ctr-nbr(3), nbr(3), dist(1)].
__global__ void lse_cat_kernel(const float* __restrict__ coords, int N_all,
                               const int* __restrict__ idx,
                               const float* __restrict__ dist,
                               float* __restrict__ cat, int n, long long total)
{
    const long long i = (long long)blockIdx.x * 256 + threadIdx.x;
    if (i >= total) return;
    const long long bq = i >> 4;
    const int b = (int)(bq / n);
    const int q = (int)(bq % n);
    const float* cb = coords + (long long)b * N_all * 3;
    const int nb = idx[i];
    const float cx = cb[q * 3 + 0], cy = cb[q * 3 + 1], cz = cb[q * 3 + 2];
    const float nx = cb[nb * 3 + 0], ny = cb[nb * 3 + 1], nz = cb[nb * 3 + 2];
    cat[0 * total + i] = cx;      cat[1 * total + i] = cy;      cat[2 * total + i] = cz;
    cat[3 * total + i] = cx - nx; cat[4 * total + i] = cy - ny; cat[5 * total + i] = cz - nz;
    cat[6 * total + i] = nx;      cat[7 * total + i] = ny;      cat[8 * total + i] = nz;
    cat[9 * total + i] = dist[i];
}

// dst[c][(bq)*K + j] = f[c][bq]
__global__ void broadcast_kernel(const float* __restrict__ f,
                                 float* __restrict__ dst,
                                 long long P1, long long total)
{
    const long long i = (long long)blockIdx.x * 256 + threadIdx.x;
    if (i >= total) return;
    const long long PK = P1 * 16;
    const long long c = i / PK;
    const long long r = i % PK;
    dst[c * PK + r] = f[c * P1 + (r >> 4)];
}

// Fused softmax(axis=K) + weighted sum: out[o][bq] = sum_j softmax(s)_j * lse_j
__global__ void attpool_kernel(const float* __restrict__ s,
                               const float* __restrict__ lse,
                               float* __restrict__ out,
                               long long P1, long long total)
{
    const long long i = (long long)blockIdx.x * 256 + threadIdx.x;
    if (i >= total) return;
    const long long o = i / P1;
    const long long bq = i % P1;
    const long long base = o * (P1 * 16) + bq * 16;
    float mx = -3.4e38f;
#pragma unroll
    for (int j = 0; j < 16; ++j) mx = fmaxf(mx, s[base + j]);
    float e[16]; float sum = 0.f;
#pragma unroll
    for (int j = 0; j < 16; ++j) { e[j] = __expf(s[base + j] - mx); sum += e[j]; }
    const float inv = 1.0f / sum;
    float acc = 0.f;
#pragma unroll
    for (int j = 0; j < 16; ++j) acc += e[j] * inv * lse[base + j];
    out[o * P1 + bq] = acc;
}

__global__ void downsample_kernel(const float* __restrict__ x,
                                  float* __restrict__ y, int n, int nn,
                                  long long total)
{
    const long long i = (long long)blockIdx.x * 256 + threadIdx.x;
    if (i >= total) return;
    const long long Pn = (long long)2 * nn; // B = 2
    const long long c = i / Pn;
    const long long r = i % Pn;
    const long long b = r / nn, q = r % nn;
    y[c * Pn + r] = x[c * ((long long)2 * n) + b * n + q];
}

// Inverse-distance-weighted 3-NN upsample, ReLU folded in (decoder concat).
__global__ void upsample_kernel(const float* __restrict__ x,
                                const int* __restrict__ idx,
                                const float* __restrict__ dist,
                                float* __restrict__ y, int nd, int nu,
                                long long total)
{
    const long long i = (long long)blockIdx.x * 256 + threadIdx.x;
    if (i >= total) return;
    const long long Pu = (long long)2 * nu;
    const long long c = i / Pu;
    const long long r = i % Pu;
    const long long b = r / nu, q = r % nu;
    const long long base = (b * nu + q) * 3;
    const float w0 = 1.0f / (dist[base + 0] + 1e-8f);
    const float w1 = 1.0f / (dist[base + 1] + 1e-8f);
    const float w2 = 1.0f / (dist[base + 2] + 1e-8f);
    const float inv = 1.0f / (w0 + w1 + w2);
    const float* xc = x + c * ((long long)2 * nd) + b * nd;
    float acc = (xc[idx[base + 0]] * w0 + xc[idx[base + 1]] * w1 +
                 xc[idx[base + 2]] * w2) * inv;
    y[c * Pu + r] = fmaxf(acc, 0.0f);
}

__global__ void copyrelu_kernel(const float* __restrict__ x,
                                float* __restrict__ y, long long total)
{
    const long long i = (long long)blockIdx.x * 256 + threadIdx.x;
    if (i >= total) return;
    y[i] = fmaxf(x[i], 0.0f);
}

// ---------------------------------------------------------------------------
// Host orchestration.
// ---------------------------------------------------------------------------
struct PackedW { const _Float16* p; int O; int nch; };

static inline int grid1d(long long n) { return (int)((n + 255) / 256); }

static void launch_gemm(hipStream_t st, const PackedW& pw, const float* b,
                        const float* g, const float* be, const float* X,
                        const float* R, float* Y, long long P,
                        int relu, int out_mode, int Nper)
{
    const long long tilesP = P >> 4;
    const int tilesO = (pw.O + 15) / 16;
    if ((tilesO & 3) == 0) {
        const long long total = (long long)(tilesO / 4) * tilesP;
        wmma_gemm_kernel<4><<<(int)((total + 7) / 8), 256, 0, st>>>(
            pw.p, b, g, be, X, R, Y, pw.O, pw.nch, P, relu, out_mode, Nper);
    } else if ((tilesO & 1) == 0) {
        const long long total = (long long)(tilesO / 2) * tilesP;
        wmma_gemm_kernel<2><<<(int)((total + 7) / 8), 256, 0, st>>>(
            pw.p, b, g, be, X, R, Y, pw.O, pw.nch, P, relu, out_mode, Nper);
    } else {
        const long long total = (long long)tilesO * tilesP;
        wmma_gemm_kernel<1><<<(int)((total + 7) / 8), 256, 0, st>>>(
            pw.p, b, g, be, X, R, Y, pw.O, pw.nch, P, relu, out_mode, Nper);
    }
}

extern "C" void kernel_launch(void* const* d_in, const int* in_sizes, int n_in,
                              void* d_out, int out_size, void* d_ws, size_t ws_size,
                              hipStream_t stream)
{
    constexpr int B = 2, N = 8192, K = 16, DR = 4, NCLS = 13;
    const float* input = (const float*)d_in[0];

    // ---- param leaves in JAX pytree (sorted-key) order ----
    int cur = 1;
    auto nextp = [&]() -> const float* { return (const float*)d_in[cur++]; };
    struct Conv { const float* b; const float* w; };
    struct Bn   { const float* be; const float* g; };
    struct ConvBn { Bn bn; Conv c; };
    auto rdConv = [&]() { Conv c; c.b = nextp(); c.w = nextp(); return c; };
    auto rdBn   = [&]() { Bn b; b.be = nextp(); b.g = nextp(); return b; };
    auto rdCB   = [&]() { ConvBn cb; cb.bn = rdBn(); cb.c = rdConv(); return cb; };

    ConvBn dec[4];
    for (int i = 0; i < 4; ++i) dec[i] = rdCB();                    // 'dec'
    struct Enc { ConvBn lse1, lse2; Conv mlp1, mlp2;
                 ConvBn p1out; Conv p1score; ConvBn p2out; Conv p2score;
                 ConvBn shrt; };
    Enc enc[4];                                                     // 'enc'
    for (int i = 0; i < 4; ++i) {
        enc[i].lse1 = rdCB(); enc[i].lse2 = rdCB();
        enc[i].mlp1 = rdConv(); enc[i].mlp2 = rdConv();
        enc[i].p1out = rdCB(); enc[i].p1score = rdConv();           // out < score
        enc[i].p2out = rdCB(); enc[i].p2score = rdConv();
        enc[i].shrt = rdCB();
    }
    ConvBn fce0 = rdCB(), fce1 = rdCB(); Conv fce2 = rdConv();      // 'fc_end'
    ConvBn fcs = rdCB();                                            // 'fc_start'
    Conv mid = rdConv();                                            // 'mid'

    // ---- workspace bump allocator ----
    char* wbase = (char*)d_ws;
    size_t woff = 0;
    auto allocb = [&](size_t bytes) -> void* {
        void* p = wbase + woff;
        woff += (bytes + 255) & ~(size_t)255;
        return p;
    };
    auto allocf = [&](long long n) { return (float*)allocb((size_t)n * 4); };
    auto alloci = [&](long long n) { return (int*)allocb((size_t)n * 4); };
    auto zfill = [&](float* p, long long n) {
        if (n > 0) zerofill_kernel<<<grid1d(n), 256, 0, stream>>>(p, n);
    };

    // persistent region: activations + packed weights
    float* rmax   = allocf(B * 3);
    float* coords = allocf((long long)B * N * 3);
    float* xin    = allocf((long long)32 * B * N);   // padded to 32 rows
    float* XA     = allocf((long long)64 * B * N);
    float* XB     = allocf((long long)64 * B * N);
    const int chS[4] = {32, 128, 256, 512};
    const int nS[4]  = {8192, 2048, 512, 128};
    float* stackb[4];
    for (int i = 0; i < 4; ++i) stackb[i] = allocf((long long)chS[i] * B * nS[i]);

    auto packW = [&](const float* W, int O, int C) -> PackedW {
        PackedW pw; pw.O = O; pw.nch = (C + 31) / 32;
        const int tilesO = (O + 15) / 16;
        const long long total = (long long)tilesO * pw.nch * 512;
        _Float16* dst = (_Float16*)allocb((size_t)total * 2);
        pack_w_kernel<<<grid1d(total), 256, 0, stream>>>(W, dst, O, C, pw.nch, total);
        pw.p = dst;
        return pw;
    };

    PackedW pw_fcs = packW(fcs.c.w, 8, 4);
    PackedW pw_mlp1[4], pw_lse1[4], pw_lse2[4], pw_p1s[4], pw_p1o[4],
            pw_p2s[4], pw_p2o[4], pw_shrt[4], pw_mlp2[4];
    const int di_arr[4] = {8, 32, 128, 256};
    const int do_arr[4] = {16, 64, 128, 256};
    for (int i = 0; i < 4; ++i) {
        const int di = di_arr[i], dd = do_arr[i], dh = dd / 2, d2o = 2 * dd;
        pw_mlp1[i] = packW(enc[i].mlp1.w, dh, di);
        pw_lse1[i] = packW(enc[i].lse1.c.w, dh, 10);
        pw_lse2[i] = packW(enc[i].lse2.c.w, dh, 10);
        pw_p1s[i]  = packW(enc[i].p1score.w, dd, dd);
        pw_p1o[i]  = packW(enc[i].p1out.c.w, dh, dd);
        pw_p2s[i]  = packW(enc[i].p2score.w, dd, dd);
        pw_p2o[i]  = packW(enc[i].p2out.c.w, dd, dd);
        pw_shrt[i] = packW(enc[i].shrt.c.w, d2o, di);
        pw_mlp2[i] = packW(enc[i].mlp2.w, d2o, dd);
    }
    PackedW pw_mid = packW(mid.w, 512, 512);
    const int decdi[4] = {1024, 512, 256, 64};
    const int decdo[4] = {256, 128, 32, 8};
    PackedW pw_dec[4];
    for (int i = 0; i < 4; ++i) pw_dec[i] = packW(dec[i].c.w, decdo[i], decdi[i]);
    PackedW pw_fce0 = packW(fce0.c.w, 64, 8);
    PackedW pw_fce1 = packW(fce1.c.w, 32, 64);
    PackedW pw_fce2 = packW(fce2.w, NCLS, 32);

    const size_t perm_off = woff;

    // ---- prep + fc_start ----
    reduce_absmax_kernel<<<B, 256, 0, stream>>>(input, rmax, N);
    {
        const long long t = (long long)B * N;
        normalize_kernel<<<grid1d(t), 256, 0, stream>>>(input, rmax, coords, xin, N, t);
        zfill(xin + 4 * t, 28 * t);                         // pad rows 4..31
    }
    launch_gemm(stream, pw_fcs, fcs.c.b, fcs.bn.g, fcs.bn.be, xin, nullptr,
                XA, (long long)B * N, 1, 0, 0);
    zfill(XA + (long long)8 * B * N, (long long)24 * B * N); // pad rows 8..31

    float* x = XA; float* xalt = XB;
    int n = N;

    // ---- encoder ----
    for (int s = 0; s < 4; ++s) {
        woff = perm_off;
        const Enc& e = enc[s];
        const int dd = do_arr[s], dh = dd / 2, d2o = 2 * dd;
        const int ddp = (dd < 32) ? 32 : dd;   // padded channel count
        const long long P1 = (long long)B * n;
        const long long PK = P1 * K;
        int*   idx  = alloci(PK);
        float* dist = allocf(PK);
        float* f    = allocf((long long)dh * P1);
        float* cat  = allocf((long long)32 * PK);
        float* lse  = allocf((long long)ddp * PK);
        float* sc   = allocf((long long)dd * PK);
        float* fp   = allocf((long long)ddp * P1);
        float* f2   = allocf((long long)dh * P1);
        float* g    = allocf((long long)ddp * P1);
        float* S    = allocf((long long)d2o * P1);

        if (dd < 32) {                         // stage-1 channel padding
            zfill(lse + (long long)dd * PK, (long long)(32 - dd) * PK);
            zfill(fp  + (long long)dd * P1, (long long)(32 - dd) * P1);
            zfill(g   + (long long)dd * P1, (long long)(32 - dd) * P1);
        }

        dim3 kg((n + 255) / 256, B);
        knn_kernel<16><<<kg, 256, 0, stream>>>(coords, N, n, n, idx, dist);

        launch_gemm(stream, pw_mlp1[s], e.mlp1.b, nullptr, nullptr, x, nullptr,
                    f, P1, 1, 0, 0);
        lse_cat_kernel<<<grid1d(PK), 256, 0, stream>>>(coords, N, idx, dist, cat, n, PK);
        zfill(cat + 10 * PK, 22 * PK);         // pad rows 10..31

        // LSE1 + attentive pool 1
        launch_gemm(stream, pw_lse1[s], e.lse1.c.b, e.lse1.bn.g, e.lse1.bn.be,
                    cat, nullptr, lse, PK, 1, 0, 0);
        broadcast_kernel<<<grid1d((long long)dh * PK), 256, 0, stream>>>(
            f, lse + (long long)dh * PK, P1, (long long)dh * PK);
        launch_gemm(stream, pw_p1s[s], e.p1score.b, nullptr, nullptr, lse,
                    nullptr, sc, PK, 0, 0, 0);
        attpool_kernel<<<grid1d((long long)dd * P1), 256, 0, stream>>>(
            sc, lse, fp, P1, (long long)dd * P1);
        launch_gemm(stream, pw_p1o[s], e.p1out.c.b, e.p1out.bn.g, e.p1out.bn.be,
                    fp, nullptr, f2, P1, 1, 0, 0);

        // LSE2 (geometric cat identical -> reuse) + attentive pool 2
        launch_gemm(stream, pw_lse2[s], e.lse2.c.b, e.lse2.bn.g, e.lse2.bn.be,
                    cat, nullptr, lse, PK, 1, 0, 0);
        broadcast_kernel<<<grid1d((long long)dh * PK), 256, 0, stream>>>(
            f2, lse + (long long)dh * PK, P1, (long long)dh * PK);
        launch_gemm(stream, pw_p2s[s], e.p2score.b, nullptr, nullptr, lse,
                    nullptr, sc, PK, 0, 0, 0);
        attpool_kernel<<<grid1d((long long)dd * P1), 256, 0, stream>>>(
            sc, lse, fp, P1, (long long)dd * P1);
        launch_gemm(stream, pw_p2o[s], e.p2out.c.b, e.p2out.bn.g, e.p2out.bn.be,
                    fp, nullptr, g, P1, 1, 0, 0);

        // residual: relu( mlp2(g) + bn_relu(short(x)) ) -> stack buffer
        launch_gemm(stream, pw_shrt[s], e.shrt.c.b, e.shrt.bn.g, e.shrt.bn.be,
                    x, nullptr, S, P1, 1, 0, 0);
        launch_gemm(stream, pw_mlp2[s], e.mlp2.b, nullptr, nullptr, g, S,
                    stackb[s], P1, 1, 0, 0);

        const int nn = n / DR;
        downsample_kernel<<<grid1d((long long)d2o * B * nn), 256, 0, stream>>>(
            stackb[s], xalt, n, nn, (long long)d2o * B * nn);
        x = xalt; xalt = (x == XA) ? XB : XA;
        n = nn;
    }

    // ---- mid ----
    woff = perm_off;
    launch_gemm(stream, pw_mid, mid.b, nullptr, nullptr, x, nullptr, xalt,
                (long long)B * n, 1, 0, 0);
    x = xalt; xalt = (x == XA) ? XB : XA;

    // ---- decoder ----
    int xch = 512;
    for (int s = 0; s < 4; ++s) {
        woff = perm_off;
        const int nd = n, nu = n * DR;
        const long long Pu = (long long)B * nu;
        int*   idx  = alloci(Pu * 3);
        float* dist = allocf(Pu * 3);
        float* cc   = allocf((long long)decdi[s] * Pu);

        dim3 kg((nu + 255) / 256, B);
        knn_kernel<3><<<kg, 256, 0, stream>>>(coords, N, nd, nu, idx, dist);
        upsample_kernel<<<grid1d((long long)xch * Pu), 256, 0, stream>>>(
            x, idx, dist, cc, nd, nu, (long long)xch * Pu);
        const int sch = chS[3 - s];
        copyrelu_kernel<<<grid1d((long long)sch * Pu), 256, 0, stream>>>(
            stackb[3 - s], cc + (long long)xch * Pu, (long long)sch * Pu);
        launch_gemm(stream, pw_dec[s], dec[s].c.b, dec[s].bn.g, dec[s].bn.be,
                    cc, nullptr, xalt, Pu, 1, 0, 0);
        x = xalt; xalt = (x == XA) ? XB : XA;
        xch = decdo[s];
        n = nu;
    }

    // ---- fc_end ----
    woff = perm_off;
    zfill(x + (long long)8 * B * N, (long long)24 * B * N);  // pad 8-ch input
    float* t1 = allocf((long long)64 * B * N);
    float* t2 = allocf((long long)32 * B * N);
    launch_gemm(stream, pw_fce0, fce0.c.b, fce0.bn.g, fce0.bn.be, x, nullptr,
                t1, (long long)B * N, 1, 0, 0);
    launch_gemm(stream, pw_fce1, fce1.c.b, fce1.bn.g, fce1.bn.be, t1, nullptr,
                t2, (long long)B * N, 1, 0, 0);
    launch_gemm(stream, pw_fce2, fce2.b, nullptr, nullptr, t2, nullptr,
                (float*)d_out, (long long)B * N, 0, 1, N);

    (void)in_sizes; (void)n_in; (void)out_size; (void)ws_size;
}
